// PAM_78185584656685
// MI455X (gfx1250) — matrix operand order repaired and checked by reading, not compiled, not run on hardware
//
#include <hip/hip_runtime.h>

#define B_   8
#define N_   4096
#define C_   512
#define CR_  64

typedef __bf16 bf16_t;
typedef __attribute__((ext_vector_type(16))) __bf16 v16bf;
typedef __attribute__((ext_vector_type(8)))  float  v8f;

struct BF16x16Raw { uint4 lo, hi; };

static __device__ __forceinline__ v16bf ld16_contig(const bf16_t* p) {
  BF16x16Raw t;
  t.lo = *(const uint4*)(p);
  t.hi = *(const uint4*)(p + 8);
  return __builtin_bit_cast(v16bf, t);
}
static __device__ __forceinline__ v16bf ld16_split(const bf16_t* p0, const bf16_t* p1) {
  BF16x16Raw t;
  t.lo = *(const uint4*)(p0);
  t.hi = *(const uint4*)(p1);
  return __builtin_bit_cast(v16bf, t);
}
static __device__ __forceinline__ v8f wmma_bf16(v16bf a, v16bf b, v8f c) {
  // D = A(16x32 bf16) * B(32x16 bf16) + C(16x16 f32)
  return __builtin_amdgcn_wmma_f32_16x16x32_bf16(false, a, false, b, (short)0, c,
                                                 false, false);
}

// ---------------------------------------------------------------------------
// Kernel 1: convert + transpose weights to bf16: WT[cout][cin]
// ---------------------------------------------------------------------------
__global__ void pam_prep_w(const float* __restrict__ wq, const float* __restrict__ wk,
                           const float* __restrict__ wv,
                           bf16_t* __restrict__ wqT, bf16_t* __restrict__ wkT,
                           bf16_t* __restrict__ wvT) {
  int idx = blockIdx.x * blockDim.x + threadIdx.x;
  if (idx < C_ * CR_) {
    int cin = idx / CR_, co = idx % CR_;
    wqT[co * C_ + cin] = (__bf16)wq[idx];
    wkT[co * C_ + cin] = (__bf16)wk[idx];
  }
  if (idx < C_ * C_) {
    int cin = idx / C_, co = idx % C_;
    wvT[co * C_ + cin] = (__bf16)wv[idx];
  }
}

// ---------------------------------------------------------------------------
// Kernel 2: fused QKV projection GEMM ([32768x512] x [512x640] in bf16).
// Out cols 0..63 -> Q[row][64], 64..127 -> K[row][64], 128..639 -> Vt[b][cout][n].
// ---------------------------------------------------------------------------
__global__ __launch_bounds__(256)
void pam_qkv(const float* __restrict__ x,
             const bf16_t* __restrict__ wqT, const bf16_t* __restrict__ wkT,
             const bf16_t* __restrict__ wvT,
             bf16_t* __restrict__ Qb, bf16_t* __restrict__ Kb,
             bf16_t* __restrict__ Vt) {
  const int lane = threadIdx.x & 31;
  const int wave = threadIdx.x >> 5;
  const int half = lane >> 4;
  const int l16  = lane & 15;
  const int mtile = blockIdx.x;                 // 16 rows of B*N
  const int ct    = blockIdx.y * 8 + wave;      // col tile 0..39
  const int c0    = ct * 16;

  const bf16_t* WT; int co;
  if (c0 < CR_)            { WT = wqT; co = c0; }
  else if (c0 < 2 * CR_)   { WT = wkT; co = c0 - CR_; }
  else                     { WT = wvT; co = c0 - 2 * CR_; }

  const float*  xrow = x  + (size_t)(mtile * 16 + l16) * C_;       // A row (M = L%16)
  const bf16_t* wrow = WT + (size_t)(co + l16) * C_;               // B col (N = L%16)

  v8f acc = {};
  for (int kk = 0; kk < C_; kk += 32) {
    // A operand: elements 0..7 = K(kk+8h..), 8..15 = K(kk+16+8h..)
    const float* pa0 = xrow + kk + 8 * half;
    const float* pa1 = pa0 + 16;
    v16bf a;
#pragma unroll
    for (int i = 0; i < 8; i++) a[i] = (__bf16)pa0[i];
#pragma unroll
    for (int i = 0; i < 8; i++) a[8 + i] = (__bf16)pa1[i];
    // B operand: elements 0..7 = K(kk+8h..), 8..15 = K(kk+16+8h..) of column (co+l16)
    v16bf b = ld16_split(wrow + kk + 8 * half, wrow + kk + 16 + 8 * half);
    acc = wmma_bf16(a, b, acc);
  }

  // D layout: VGPR r -> row (r + 8*half), col = l16
  if (c0 < 2 * CR_) {
    bf16_t* out = (c0 < CR_) ? Qb : Kb;
#pragma unroll
    for (int r = 0; r < 8; r++) {
      int rr = mtile * 16 + r + 8 * half;                 // global row in [0, B*N)
      out[(size_t)rr * CR_ + co + l16] = (__bf16)acc[r];
    }
  } else {
    int base = mtile * 16;
    int bb   = base >> 12;                                // batch (N_=4096 per batch)
#pragma unroll
    for (int r = 0; r < 8; r++) {
      int rr = base + r + 8 * half;
      int n  = rr & (N_ - 1);
      Vt[((size_t)bb * C_ + co + l16) * N_ + n] = (__bf16)acc[r];
    }
  }
}

// ---------------------------------------------------------------------------
// Kernel 3: flash attention + gamma residual.
// Grid (256 query tiles, 8 batches); block = 128 threads = 4 waves.
// K tiles (64 keys x 64 feats bf16 = 8 KB) are double-buffered in LDS via
// gfx1250 async global->LDS copies (ASYNCcnt), shared by all 4 waves.
// Each wave: 16 query rows x 128 V-cols (8 f32 accum tiles).
// ---------------------------------------------------------------------------
__global__ __launch_bounds__(128)
void pam_attn(const float* __restrict__ x,
              const bf16_t* __restrict__ Qb, const bf16_t* __restrict__ Kb,
              const bf16_t* __restrict__ Vt, const float* __restrict__ gammap,
              float* __restrict__ out) {
  __shared__ __align__(16) bf16_t Klds[2][64 * 64];   // 2 x 8 KB K tiles
  __shared__ __align__(16) bf16_t Plds[4][16 * 64];   // per-wave private P tile

  const int tid   = threadIdx.x;
  const int lane  = tid & 31;
  const int wave  = tid >> 5;
  const int half  = lane >> 4;
  const int l16   = lane & 15;
  const int qt    = blockIdx.x;
  const int b     = blockIdx.y;
  const int qbase = qt * 16;
  const int cbase = wave * 128;
  const float gamma = gammap[0];

  const char* kbatch = (const char*)(Kb + (size_t)b * N_ * CR_);  // batch K base
  const unsigned lds0 = (unsigned)(uintptr_t)(&Klds[0][0]) + (unsigned)(tid * 16);
  const unsigned lds1 = (unsigned)(uintptr_t)(&Klds[1][0]) + (unsigned)(tid * 16);

  // async prefetch of K tile 0 (each thread: 4 x 16B, instruction offset is
  // added to BOTH the global and the LDS address per the TDM/async spec)
  {
    const char* src = kbatch + tid * 16;
    asm volatile("global_load_async_to_lds_b128 %0, %1, off"              :: "v"(lds0), "v"(src) : "memory");
    asm volatile("global_load_async_to_lds_b128 %0, %1, off offset:2048"  :: "v"(lds0), "v"(src) : "memory");
    asm volatile("global_load_async_to_lds_b128 %0, %1, off offset:4096"  :: "v"(lds0), "v"(src) : "memory");
    asm volatile("global_load_async_to_lds_b128 %0, %1, off offset:6144"  :: "v"(lds0), "v"(src) : "memory");
  }

  // Q as A operands for the two 32-wide k chunks of the 64-dim dot product
  const bf16_t* qrow = Qb + ((size_t)(b * N_ + qbase + l16)) * CR_;
  v16bf qa[2];
#pragma unroll
  for (int kc = 0; kc < 2; kc++)
    qa[kc] = ld16_split(qrow + kc * 32 + 8 * half,
                        qrow + kc * 32 + 16 + 8 * half);

  v8f o[8] = {};
  float m_run[8], l_run[8];
#pragma unroll
  for (int r = 0; r < 8; r++) { m_run[r] = -1.0e30f; l_run[r] = 0.0f; }

  for (int i = 0; i < N_ / 64; i++) {
    const int kb  = i * 64;
    const int cur = i & 1;

    // my async copies done -> barrier -> whole tile visible to all waves
    asm volatile("s_wait_asynccnt 0x0" ::: "memory");
    __syncthreads();

    // kick off next K tile into the other buffer (safe: its last readers
    // finished before the barrier above)
    if (i < N_ / 64 - 1) {
      const char* src = kbatch + (size_t)(kb + 64) * CR_ * sizeof(bf16_t) + tid * 16;
      const unsigned dst = cur ? lds0 : lds1;
      asm volatile("global_load_async_to_lds_b128 %0, %1, off"              :: "v"(dst), "v"(src) : "memory");
      asm volatile("global_load_async_to_lds_b128 %0, %1, off offset:2048"  :: "v"(dst), "v"(src) : "memory");
      asm volatile("global_load_async_to_lds_b128 %0, %1, off offset:4096"  :: "v"(dst), "v"(src) : "memory");
      asm volatile("global_load_async_to_lds_b128 %0, %1, off offset:6144"  :: "v"(dst), "v"(src) : "memory");
    }

    // ---- S = Q @ K^T for 64 keys (K tile from LDS): four 16x16 tiles ---
    v8f s[4];
#pragma unroll
    for (int t = 0; t < 4; t++) {
      const bf16_t* krow = &Klds[cur][(t * 16 + l16) * CR_];
      v8f c = {};
#pragma unroll
      for (int kc = 0; kc < 2; kc++)
        c = wmma_bf16(qa[kc], ld16_split(krow + kc * 32 + 8 * half, krow + kc * 32 + 16 + 8 * half), c);
      s[t] = c;
    }

    // ---- online softmax ----------------------------------------------
    float rmax[8];
#pragma unroll
    for (int r = 0; r < 8; r++)
      rmax[r] = fmaxf(fmaxf(s[0][r], s[1][r]), fmaxf(s[2][r], s[3][r]));
#pragma unroll
    for (int mask = 1; mask <= 8; mask <<= 1)
#pragma unroll
      for (int r = 0; r < 8; r++)
        rmax[r] = fmaxf(rmax[r], __shfl_xor(rmax[r], mask, 32));

    float alpha[8], rsum[8];
#pragma unroll
    for (int r = 0; r < 8; r++) {
      float mnew = fmaxf(m_run[r], rmax[r]);
      alpha[r]   = __expf(m_run[r] - mnew);
      m_run[r]   = mnew;
      rsum[r]    = 0.0f;
    }
    // p = exp(s - m), stash bf16 P into this wave's private LDS tile
#pragma unroll
    for (int t = 0; t < 4; t++)
#pragma unroll
      for (int r = 0; r < 8; r++) {
        float p = __expf(s[t][r] - m_run[r]);
        rsum[r] += p;
        Plds[wave][(r + 8 * half) * 64 + t * 16 + l16] = (__bf16)p;
      }
#pragma unroll
    for (int mask = 1; mask <= 8; mask <<= 1)
#pragma unroll
      for (int r = 0; r < 8; r++)
        rsum[r] += __shfl_xor(rsum[r], mask, 32);
#pragma unroll
    for (int r = 0; r < 8; r++)
      l_run[r] = l_run[r] * alpha[r] + rsum[r];
    // rescale accumulators (row scale matches D-layout row per VGPR)
#pragma unroll
    for (int t = 0; t < 8; t++)
#pragma unroll
      for (int r = 0; r < 8; r++)
        o[t][r] *= alpha[r];

    // wave-synchronous LDS RAW on the private P tile
    asm volatile("s_wait_dscnt 0" ::: "memory");
    __builtin_amdgcn_wave_barrier();

    const bf16_t* prow = &Plds[wave][l16 * 64];
    v16bf pa[2];
#pragma unroll
    for (int kc = 0; kc < 2; kc++)
      pa[kc] = ld16_split(prow + kc * 32 + 8 * half,
                          prow + kc * 32 + 16 + 8 * half);

    // ---- O += P @ V over this wave's 128 columns ---------------------
#pragma unroll
    for (int t = 0; t < 8; t++) {
      const bf16_t* vrow = Vt + ((size_t)(b * C_ + cbase + t * 16 + l16)) * N_ + kb;
#pragma unroll
      for (int kc = 0; kc < 2; kc++)
        o[t] = wmma_bf16(pa[kc], ld16_split(vrow + kc * 32 + 8 * half, vrow + kc * 32 + 16 + 8 * half), o[t]);
    }
  }

  // ---- epilogue: out = gamma * O / l + x ------------------------------
  float ginv[8];
#pragma unroll
  for (int r = 0; r < 8; r++) ginv[r] = gamma / l_run[r];
#pragma unroll
  for (int t = 0; t < 8; t++)
#pragma unroll
    for (int r = 0; r < 8; r++) {
      int row = qbase + r + 8 * half;
      int col = cbase + t * 16 + l16;
      size_t idx = ((size_t)(b * N_ + row)) * C_ + col;
      out[idx] = o[t][r] * ginv[r] + x[idx];
    }
}

// ---------------------------------------------------------------------------
extern "C" void kernel_launch(void* const* d_in, const int* in_sizes, int n_in,
                              void* d_out, int out_size, void* d_ws, size_t ws_size,
                              hipStream_t stream) {
  const float* x     = (const float*)d_in[0];
  const float* wq    = (const float*)d_in[1];
  const float* wk    = (const float*)d_in[2];
  const float* wv    = (const float*)d_in[3];
  const float* gamma = (const float*)d_in[4];
  float* out = (float*)d_out;

  char* ws = (char*)d_ws;
  bf16_t* wqT = (bf16_t*)(ws);                                  //  64 KB
  bf16_t* wkT = (bf16_t*)(ws + (64 << 10));                     //  64 KB
  bf16_t* wvT = (bf16_t*)(ws + (128 << 10));                    // 512 KB
  bf16_t* Qb  = (bf16_t*)(ws + (640 << 10));                    //   4 MB
  bf16_t* Kb  = (bf16_t*)(ws + (640 << 10) + (4 << 20));        //   4 MB
  bf16_t* Vt  = (bf16_t*)(ws + (640 << 10) + (8 << 20));        //  32 MB

  // 1) weights -> bf16, transposed
  pam_prep_w<<<(C_ * C_ + 255) / 256, 256, 0, stream>>>(wq, wk, wv, wqT, wkT, wvT);

  // 2) fused QKV projection: 2048 row-tiles x 5 col-groups (8 waves each)
  pam_qkv<<<dim3((B_ * N_) / 16, 5), 256, 0, stream>>>(x, wqT, wkT, wvT, Qb, Kb, Vt);

  // 3) flash attention (async K staging) + residual
  pam_attn<<<dim3(N_ / 16, B_), 128, 0, stream>>>(x, Qb, Kb, Vt, gamma, out);
}
